// VidSumGNN_8203387536001
// MI455X (gfx1250) — matrix operand order repaired and checked by reading, not compiled, not run on hardware
//
#include <hip/hip_runtime.h>
#include <hip/hip_bf16.h>
#include <math.h>

// ---------------------------------------------------------------------------
// Problem constants (match reference)
// ---------------------------------------------------------------------------
#define GN   16384      // nodes
#define GE   65536      // edges (before self loops)
#define GEE  (GE + GN)  // edges + self loops = 81920
#define GIN  1536
#define GHID 1024
#define GHEADS 8
#define GHD  128

// GEMM tiling: block = 256 threads = 8 waves -> 128x128 C tile.
// wave (wm 0..3, wn 0..1) owns 32x64 as 2x4 v_wmma_f32_16x16x32_bf16 tiles.
#define TM 128
#define TN 128
#define TK 32

typedef __attribute__((ext_vector_type(16))) __bf16 bf16x16;
typedef __attribute__((ext_vector_type(8)))  float  f32x8;
typedef __attribute__((ext_vector_type(4)))  int    intx4;

#define GLOBAL_AS __attribute__((address_space(1)))
#define LDS_AS    __attribute__((address_space(3)))

union FragAB { bf16x16 v; unsigned int u[8]; };
union FragC  { f32x8  v; float f[8]; };

// CDNA5 async global->LDS (ASYNCcnt-tracked) when the toolchain exposes it.
#if defined(__has_builtin)
#if __has_builtin(__builtin_amdgcn_global_load_async_to_lds_b128) && \
    __has_builtin(__builtin_amdgcn_s_wait_asynccnt)
#define USE_ASYNC_LDS 1
#endif
#endif
#ifndef USE_ASYNC_LDS
#define USE_ASYNC_LDS 0
#endif

// ---------------------------------------------------------------------------
// Helpers
// ---------------------------------------------------------------------------
static __device__ __forceinline__ unsigned short f2bf(float f) {
    unsigned u = __float_as_uint(f);
    unsigned r = u + 0x7FFFu + ((u >> 16) & 1u);   // round-to-nearest-even
    return (unsigned short)(r >> 16);
}

static __device__ __forceinline__ int ordered_int(float f) {
    int i = __float_as_int(f);
    return (i >= 0) ? i : (i ^ 0x7FFFFFFF);        // monotonic float->int
}

static __device__ __forceinline__ float ordered_int_inv(int i) {
    return __int_as_float((i >= 0) ? i : (i ^ 0x7FFFFFFF));
}

// ---------------------------------------------------------------------------
// fp32 -> bf16 streaming convert / fills
// ---------------------------------------------------------------------------
__global__ void f32_to_bf16_kernel(const float* __restrict__ in,
                                   unsigned short* __restrict__ out, size_t n) {
    size_t i = (size_t)blockIdx.x * blockDim.x + threadIdx.x;
    size_t stride = (size_t)gridDim.x * blockDim.x;
    for (; i < n; i += stride) out[i] = f2bf(in[i]);
}

__global__ void fill_u32_kernel(unsigned int* __restrict__ p, unsigned int v, size_t n) {
    size_t i = (size_t)blockIdx.x * blockDim.x + threadIdx.x;
    size_t stride = (size_t)gridDim.x * blockDim.x;
    for (; i < n; i += stride) p[i] = v;
}

// ---------------------------------------------------------------------------
// WMMA GEMM:  C[M,N] = A[M,K](bf16) * B[K,N](bf16) + bias, optional SiLU,
// optional bf16 mirror.
// ---------------------------------------------------------------------------
__global__ __launch_bounds__(256)
void gemm_bf16_wmma_kernel(const unsigned short* __restrict__ A,
                           const unsigned short* __restrict__ B,
                           const float* __restrict__ bias,
                           float* __restrict__ C,
                           unsigned short* __restrict__ Cb,
                           int M, int Nn, int K, int act) {
    __shared__ unsigned short As[TM * TK];   // [row][k] row-major, 8 KB
    __shared__ unsigned short Bs[TN * TK];   // [col][k] K-contig per column, 8 KB

    const int tid  = threadIdx.x;
    const int lane = tid & 31;
    const int wid  = tid >> 5;
    const int wm   = wid & 3;      // 0..3  (32-row slices)
    const int wn   = wid >> 2;     // 0..1  (64-col slices)
    const int m_blk = blockIdx.y * TM;
    const int n_blk = blockIdx.x * TN;

    FragC d[2][4];
    #pragma unroll
    for (int i = 0; i < 2; ++i)
        #pragma unroll
        for (int j = 0; j < 4; ++j)
            #pragma unroll
            for (int r = 0; r < 8; ++r) d[i][j].f[r] = 0.0f;

    for (int k0 = 0; k0 < K; k0 += TK) {
        // ---- stage A tile: 128x32 bf16, async global->LDS (layout kept) --
#if USE_ASYNC_LDS
        #pragma unroll
        for (int it = 0; it < 2; ++it) {
            int i   = tid + it * 256;       // 0..511 b128 transfers
            int row = i >> 2;
            int q   = i & 3;
            const unsigned short* gs = A + (size_t)(m_blk + row) * K + k0 + q * 8;
            unsigned short* ls = &As[row * TK + q * 8];
            __builtin_amdgcn_global_load_async_to_lds_b128(
                (GLOBAL_AS intx4*)gs, (LDS_AS intx4*)ls, 0, 0);
        }
#else
        #pragma unroll
        for (int it = 0; it < 2; ++it) {
            int i   = tid + it * 256;
            int row = i >> 2;
            int q   = i & 3;
            const uint4* src = (const uint4*)(A + (size_t)(m_blk + row) * K + k0 + q * 8);
            *(uint4*)&As[row * TK + q * 8] = *src;
        }
#endif
        // ---- stage B tile transposed: Bs[n][k] = B[k0+k][n_blk+n] --------
        // Phase 1: batch all global loads into registers (loads pipeline
        // under a single wait instead of load->wait->store x8 serialization).
        unsigned int pv[8];
        #pragma unroll
        for (int it = 0; it < 8; ++it) {
            int i  = tid + it * 256;        // 0..2047 (k, n-pair)
            int k  = i >> 6;
            int n  = (i & 63) * 2;
            pv[it] = *(const unsigned int*)(B + (size_t)(k0 + k) * Nn + n_blk + n);
        }
        // Phase 2: scatter to LDS (transpose to [col][k]).
        #pragma unroll
        for (int it = 0; it < 8; ++it) {
            int i  = tid + it * 256;
            int k  = i >> 6;
            int n  = (i & 63) * 2;
            Bs[(n    ) * TK + k] = (unsigned short)(pv[it] & 0xFFFFu);
            Bs[(n + 1) * TK + k] = (unsigned short)(pv[it] >> 16);
        }
        // prefetch next K tile while this one is consumed
        if (k0 + TK < K) {
            __builtin_prefetch(A + (size_t)(m_blk + (tid >> 1)) * K + k0 + TK, 0, 1);
            __builtin_prefetch(B + (size_t)(k0 + TK + (tid & 31)) * Nn + n_blk, 0, 1);
        }
#if USE_ASYNC_LDS
        __builtin_amdgcn_s_wait_asynccnt(0);
#endif
        __syncthreads();

        // ---- fragment loads per ISA 16-bit A 16x32 / B 32x16 layouts -----
        const int lr = lane & 15;
        const int hi = lane >> 4;
        FragAB a[2], b[4];
        #pragma unroll
        for (int i = 0; i < 2; ++i) {
            int m = wm * 32 + i * 16 + lr;
            #pragma unroll
            for (int j = 0; j < 8; ++j) {
                int kk = ((j & 4) ? 16 : 0) + hi * 8 + 2 * (j & 3);
                a[i].u[j] = *(const unsigned int*)&As[m * TK + kk];
            }
        }
        #pragma unroll
        for (int j2 = 0; j2 < 4; ++j2) {
            int n = wn * 64 + j2 * 16 + lr;
            #pragma unroll
            for (int v = 0; v < 8; ++v)
                b[j2].u[v] = *(const unsigned int*)&Bs[n * TK + ((v & 4) ? 16 : 0) + hi * 8 + 2 * (v & 3)];
        }

        #pragma unroll
        for (int i = 0; i < 2; ++i)
            #pragma unroll
            for (int j = 0; j < 4; ++j)
                d[i][j].v = __builtin_amdgcn_wmma_f32_16x16x32_bf16(
                    false, a[i].v, false, b[j].v, (short)0, d[i][j].v, false, false);

        __syncthreads();
    }

    // ---- epilogue: bias (+SiLU), write fp32 (+bf16 mirror) ---------------
    const int lr = lane & 15;
    const int hi = lane >> 4;
    #pragma unroll
    for (int i = 0; i < 2; ++i) {
        #pragma unroll
        for (int j = 0; j < 4; ++j) {
            int gn = n_blk + wn * 64 + j * 16 + lr;
            float bv = bias ? bias[gn] : 0.0f;
            #pragma unroll
            for (int r = 0; r < 8; ++r) {
                int gm = m_blk + wm * 32 + i * 16 + r + hi * 8;
                float val = d[i][j].f[r] + bv;
                if (act == 1) {                     // SiLU with fast rcp
                    float e = __expf(-val);
                    val = val * __builtin_amdgcn_rcpf(1.0f + e);
                }
                size_t o = (size_t)gm * Nn + gn;
                C[o] = val;
                if (Cb) Cb[o] = f2bf(val);
            }
        }
    }
}

// ---------------------------------------------------------------------------
// out = (residual?) + gelu(layer_norm(in + pre_bias?; g, b)); bf16 mirror opt.
// One 256-thread block per row of 1024.
// ---------------------------------------------------------------------------
__global__ __launch_bounds__(256)
void ln_gelu_kernel(const float* __restrict__ in, const float* __restrict__ pre_bias,
                    const float* __restrict__ g, const float* __restrict__ b,
                    const float* __restrict__ residual,
                    float* __restrict__ out, unsigned short* __restrict__ outb) {
    __shared__ float red[256];
    __shared__ float red2[256];
    const size_t row = blockIdx.x;
    const float* xi = in + row * GHID;
    float v[4];
    float s = 0.f, s2 = 0.f;
    #pragma unroll
    for (int i = 0; i < 4; ++i) {
        int c = threadIdx.x + i * 256;
        float x = xi[c];
        if (pre_bias) x += pre_bias[c];
        v[i] = x; s += x; s2 += x * x;
    }
    red[threadIdx.x] = s; red2[threadIdx.x] = s2;
    __syncthreads();
    for (int off = 128; off > 0; off >>= 1) {
        if (threadIdx.x < (unsigned)off) {
            red[threadIdx.x]  += red[threadIdx.x + off];
            red2[threadIdx.x] += red2[threadIdx.x + off];
        }
        __syncthreads();
    }
    const float mu  = red[0] * (1.0f / GHID);
    const float var = red2[0] * (1.0f / GHID) - mu * mu;
    const float inv = rsqrtf(var + 1e-5f);
    #pragma unroll
    for (int i = 0; i < 4; ++i) {
        int c = threadIdx.x + i * 256;
        float y  = (v[i] - mu) * inv * g[c] + b[c];
        float ge = 0.5f * y * (1.0f + erff(y * 0.70710678118654752f)); // exact GELU
        float o  = (residual ? residual[row * GHID + c] : 0.0f) + ge;
        out[row * GHID + c] = o;
        if (outb) outb[row * GHID + c] = f2bf(o);
    }
}

// ---------------------------------------------------------------------------
// GATv2 edge pass 1: logits + segment-max (ordered-int atomicMax).
// One block per edge, one wave per head, 4 channels/lane.
// ---------------------------------------------------------------------------
__global__ __launch_bounds__(256)
void edge_logits_kernel(const float* __restrict__ xl, const float* __restrict__ xr,
                        const float* __restrict__ att, const int* __restrict__ ei,
                        float* __restrict__ logit, int* __restrict__ mInt) {
    const int e = blockIdx.x;
    int src, dst;
    if (e < GE) { src = ei[e]; dst = ei[GE + e]; } else { src = e - GE; dst = src; }
    const int h    = threadIdx.x >> 5;
    const int lane = threadIdx.x & 31;
    const float* pl = xl + (size_t)src * GHID + h * GHD;
    const float* pr = xr + (size_t)dst * GHID + h * GHD;
    const float* pa = att + h * GHD;
    float sum = 0.f;
    #pragma unroll
    for (int i = 0; i < 4; ++i) {
        int c = lane + i * 32;
        float t = pl[c] + pr[c];
        t = (t > 0.f) ? t : 0.2f * t;          // leaky_relu
        sum += pa[c] * t;
    }
    #pragma unroll
    for (int off = 16; off > 0; off >>= 1) sum += __shfl_xor(sum, off, 32);
    if (lane == 0) {
        logit[(size_t)e * GHEADS + h] = sum;
        atomicMax(&mInt[dst * GHEADS + h], ordered_int(sum));
    }
}

// Pass 2: a = exp(logit - m[dst]); denom[dst] += a  (overwrites logit with a)
__global__ void edge_exp_kernel(float* __restrict__ logit, const int* __restrict__ mInt,
                                const int* __restrict__ ei, float* __restrict__ denom) {
    int idx = blockIdx.x * blockDim.x + threadIdx.x;
    if (idx >= GEE * GHEADS) return;
    int e = idx >> 3, h = idx & 7;
    int dst = (e < GE) ? ei[GE + e] : (e - GE);
    float m = ordered_int_inv(mInt[dst * GHEADS + h]);
    float a = __expf(logit[idx] - m);
    logit[idx] = a;
    atomicAdd(&denom[dst * GHEADS + h], a);
}

// Pass 3: agg[dst] += (a/denom[dst]) * xl[src].  Block per edge, 4 ch/thread.
__global__ __launch_bounds__(256)
void edge_agg_kernel(const float* __restrict__ a, const float* __restrict__ denom,
                     const float* __restrict__ xl, const int* __restrict__ ei,
                     float* __restrict__ agg) {
    const int e = blockIdx.x;
    int src, dst;
    if (e < GE) { src = ei[e]; dst = ei[GE + e]; } else { src = e - GE; dst = src; }
    const int t = threadIdx.x;
    const int h = t >> 5;
    const float alpha = a[(size_t)e * GHEADS + h] *
                        __builtin_amdgcn_rcpf(denom[dst * GHEADS + h]);
    const float* pl = xl + (size_t)src * GHID + h * GHD;
    float*       po = agg + (size_t)dst * GHID + h * GHD;
    #pragma unroll
    for (int i = 0; i < 4; ++i) {
        int c = (t & 31) + i * 32;
        atomicAdd(&po[c], alpha * pl[c]);
    }
}

// ---------------------------------------------------------------------------
// Final scorer dot: scores[r] = dot(s2[r,:128], W3) + b3. Wave per row.
// ---------------------------------------------------------------------------
__global__ __launch_bounds__(256)
void score_kernel(const float* __restrict__ s2, const float* __restrict__ W3,
                  const float* __restrict__ b3, float* __restrict__ scores) {
    const int wid  = threadIdx.x >> 5;
    const int lane = threadIdx.x & 31;
    const int row  = blockIdx.x * 8 + wid;
    if (row >= GN) return;
    const float* p = s2 + (size_t)row * 128;
    float sum = 0.f;
    #pragma unroll
    for (int i = 0; i < 4; ++i) { int c = lane + i * 32; sum += p[c] * W3[c]; }
    #pragma unroll
    for (int off = 16; off > 0; off >>= 1) sum += __shfl_xor(sum, off, 32);
    if (lane == 0) scores[row] = sum + b3[0];
}

// ---------------------------------------------------------------------------
// Host orchestration
// ---------------------------------------------------------------------------
extern "C" void kernel_launch(void* const* d_in, const int* in_sizes, int n_in,
                              void* d_out, int out_size, void* d_ws, size_t ws_size,
                              hipStream_t stream) {
    (void)in_sizes; (void)n_in; (void)out_size; (void)ws_size;

    const float* x      = (const float*)d_in[0];
    const int*   ei     = (const int*)d_in[1];
    const float* W_in   = (const float*)d_in[2];
    const float* b_in   = (const float*)d_in[3];
    const float* ln1_g  = (const float*)d_in[4];
    const float* ln1_b  = (const float*)d_in[5];
    const float* g1_Wl  = (const float*)d_in[6];
    const float* g1_bl  = (const float*)d_in[7];
    const float* g1_Wr  = (const float*)d_in[8];
    const float* g1_br  = (const float*)d_in[9];
    const float* g1_att = (const float*)d_in[10];
    const float* g1_bo  = (const float*)d_in[11];
    const float* ln2_g  = (const float*)d_in[12];
    const float* ln2_b  = (const float*)d_in[13];
    const float* g2_Wl  = (const float*)d_in[14];
    const float* g2_bl  = (const float*)d_in[15];
    const float* g2_Wr  = (const float*)d_in[16];
    const float* g2_br  = (const float*)d_in[17];
    const float* g2_att = (const float*)d_in[18];
    const float* g2_bo  = (const float*)d_in[19];
    const float* ln3_g  = (const float*)d_in[20];
    const float* ln3_b  = (const float*)d_in[21];
    const float* Ws1    = (const float*)d_in[22];
    const float* bs1    = (const float*)d_in[23];
    const float* Ws2    = (const float*)d_in[24];
    const float* bs2    = (const float*)d_in[25];
    const float* Ws3    = (const float*)d_in[26];
    const float* bs3    = (const float*)d_in[27];

    // ---- workspace layout ------------------------------------------------
    char* base = (char*)d_ws;
    size_t off = 0;
    auto alloc = [&](size_t bytes) -> void* {
        off = (off + 255) & ~(size_t)255;
        void* r = base + off;
        off += bytes;
        return r;
    };
    unsigned short* xb    = (unsigned short*)alloc((size_t)GN * GIN * 2);
    unsigned short* wWin  = (unsigned short*)alloc((size_t)GIN * GHID * 2);
    unsigned short* w1l   = (unsigned short*)alloc((size_t)GHID * GHID * 2);
    unsigned short* w1r   = (unsigned short*)alloc((size_t)GHID * GHID * 2);
    unsigned short* w2l   = (unsigned short*)alloc((size_t)GHID * GHID * 2);
    unsigned short* w2r   = (unsigned short*)alloc((size_t)GHID * GHID * 2);
    unsigned short* ws1b  = (unsigned short*)alloc((size_t)GHID * 512 * 2);
    unsigned short* ws2b  = (unsigned short*)alloc((size_t)512 * 128 * 2);
    float*          h0    = (float*)alloc((size_t)GN * GHID * 4);  // reused as s1
    float*          h     = (float*)alloc((size_t)GN * GHID * 4);
    unsigned short* hb    = (unsigned short*)alloc((size_t)GN * GHID * 2);
    float*          xl    = (float*)alloc((size_t)GN * GHID * 4);  // reused as s2
    float*          xr    = (float*)alloc((size_t)GN * GHID * 4);
    float*          agg   = (float*)alloc((size_t)GN * GHID * 4);  // reused as s1 bf16
    float*          logit = (float*)alloc((size_t)GEE * GHEADS * 4);
    int*            mInt  = (int*)alloc((size_t)GN * GHEADS * 4);
    float*          denom = (float*)alloc((size_t)GN * GHEADS * 4);

    float*          scores_out = (float*)d_out;
    float*          h_out      = (float*)d_out + GN;
    float*          s1  = h0;
    unsigned short* s1b = (unsigned short*)agg;
    float*          s2  = xl;

    const dim3 blk256(256);
    auto cvt = [&](const float* src, unsigned short* dst, size_t n) {
        int g = (int)((n + 256 * 8 - 1) / (256 * 8));
        if (g > 4096) g = 4096;
        f32_to_bf16_kernel<<<g, blk256, 0, stream>>>(src, dst, n);
    };
    auto gemm = [&](const unsigned short* A, const unsigned short* B, const float* bias,
                    float* C, unsigned short* Cb, int M, int Nn, int K, int act) {
        dim3 grid(Nn / TN, M / TM);
        gemm_bf16_wmma_kernel<<<grid, blk256, 0, stream>>>(A, B, bias, C, Cb, M, Nn, K, act);
    };
    auto gat_layer = [&](const unsigned short* hin_bf16, const float* bl_,
                         const float* br_, const float* att_,
                         unsigned short* wlbuf, unsigned short* wrbuf) {
        gemm(hin_bf16, wlbuf, bl_, xl, nullptr, GN, GHID, GHID, 0);
        gemm(hin_bf16, wrbuf, br_, xr, nullptr, GN, GHID, GHID, 0);
        fill_u32_kernel<<<512, blk256, 0, stream>>>((unsigned int*)mInt, 0x80000000u,
                                                    (size_t)GN * GHEADS);
        fill_u32_kernel<<<512, blk256, 0, stream>>>((unsigned int*)denom, 0u,
                                                    (size_t)GN * GHEADS);
        fill_u32_kernel<<<4096, blk256, 0, stream>>>((unsigned int*)agg, 0u,
                                                     (size_t)GN * GHID);
        edge_logits_kernel<<<GEE, blk256, 0, stream>>>(xl, xr, att_, ei, logit, mInt);
        edge_exp_kernel<<<(GEE * GHEADS + 255) / 256, blk256, 0, stream>>>(logit, mInt, ei, denom);
        edge_agg_kernel<<<GEE, blk256, 0, stream>>>(logit, denom, xl, ei, agg);
    };

    // ---- weight / input conversion --------------------------------------
    cvt(x,     xb,   (size_t)GN * GIN);
    cvt(W_in,  wWin, (size_t)GIN * GHID);
    cvt(g1_Wl, w1l,  (size_t)GHID * GHID);
    cvt(g1_Wr, w1r,  (size_t)GHID * GHID);
    cvt(g2_Wl, w2l,  (size_t)GHID * GHID);
    cvt(g2_Wr, w2r,  (size_t)GHID * GHID);
    cvt(Ws1,   ws1b, (size_t)GHID * 512);
    cvt(Ws2,   ws2b, (size_t)512 * 128);

    // ---- stage 1: h = gelu(ln1(x @ W_in + b_in)) -------------------------
    gemm(xb, wWin, b_in, h0, nullptr, GN, GHID, GIN, 0);
    ln_gelu_kernel<<<GN, blk256, 0, stream>>>(h0, nullptr, ln1_g, ln1_b, nullptr, h, hb);

    // ---- GAT layer 1 + residual ln/gelu ----------------------------------
    gat_layer(hb, g1_bl, g1_br, g1_att, w1l, w1r);
    ln_gelu_kernel<<<GN, blk256, 0, stream>>>(agg, g1_bo, ln2_g, ln2_b, h, h, hb);

    // ---- GAT layer 2 + residual ln/gelu (final h written to d_out) -------
    gat_layer(hb, g2_bl, g2_br, g2_att, w2l, w2r);
    ln_gelu_kernel<<<GN, blk256, 0, stream>>>(agg, g2_bo, ln3_g, ln3_b, h, h_out, hb);

    // ---- scorer MLP ------------------------------------------------------
    gemm(hb,  ws1b, bs1, s1, s1b,     GN, 512, GHID, 1);   // SiLU
    gemm(s1b, ws2b, bs2, s2, nullptr, GN, 128, 512,  1);   // SiLU
    score_kernel<<<GN / 8, blk256, 0, stream>>>(s2, Ws3, bs3, scores_out);
}
